// Attention_72902774882333
// MI455X (gfx1250) — matrix operand order repaired and checked by reading, not compiled, not run on hardware
//
#include <hip/hip_runtime.h>
#include <hip/hip_bf16.h>
#include <stdint.h>

constexpr int kB = 4, kC = 256, kN = 4096, kH = 4, kD = 64;

typedef __attribute__((ext_vector_type(16))) __bf16 bf16x16;
typedef __attribute__((ext_vector_type(8)))  float  f32x8;

struct Pair128 { uint4 a, b; };

__device__ __forceinline__ unsigned short f2bf(float f) {
  unsigned u = __builtin_bit_cast(unsigned, f);
  u += 0x7fffu + ((u >> 16) & 1u);          // round-to-nearest-even
  return (unsigned short)(u >> 16);
}
__device__ __forceinline__ unsigned pack_bf2(float a, float b) {
  return (unsigned)f2bf(a) | ((unsigned)f2bf(b) << 16);
}
// A-fragment (16x32 bf16): per-lane 16 contiguous K elements
__device__ __forceinline__ bf16x16 load_frag_a(const unsigned short* p) {
  Pair128 t;
  t.a = *reinterpret_cast<const uint4*>(p);
  t.b = *reinterpret_cast<const uint4*>(p + 8);
  return __builtin_bit_cast(bf16x16, t);
}
// B-fragment (32x16 bf16): per-lane 16 contiguous K elements
__device__ __forceinline__ bf16x16 load_frag_b(const unsigned short* p) {
  Pair128 t;
  t.a = *reinterpret_cast<const uint4*>(p);
  t.b = *reinterpret_cast<const uint4*>(p + 8);
  return __builtin_bit_cast(bf16x16, t);
}
__device__ __forceinline__ f32x8 wmma_bf16(bf16x16 a, bf16x16 b, f32x8 c) {
  return __builtin_amdgcn_wmma_f32_16x16x32_bf16(false, a, false, b, (short)0, c,
                                                 false, false);
}

// ---------------- kernel 1: weights f32 -> bf16 (wq|wk|wv|wp packed) --------
__global__ __launch_bounds__(256) void attn_cvt_w(
    const float* __restrict__ wq, const float* __restrict__ wk,
    const float* __restrict__ wv, const float* __restrict__ wp,
    unsigned short* __restrict__ out) {
  const int i = blockIdx.x * blockDim.x + threadIdx.x;
  if (i < kC * kC) {
    out[i]               = f2bf(wq[i]);
    out[i + kC * kC]     = f2bf(wk[i]);
    out[i + 2 * kC * kC] = f2bf(wv[i]);
    out[i + 3 * kC * kC] = f2bf(wp[i]);
  }
}

// ---------------- kernel 2: x [B,C,N] f32 -> xT [B,N,C] bf16 ----------------
__global__ __launch_bounds__(256) void attn_cvt_x(
    const float* __restrict__ x, unsigned short* __restrict__ xT) {
  const int nn = blockIdx.x * 32 + threadIdx.x;
  const int c  = blockIdx.y * 8 + threadIdx.y;
  const int b  = blockIdx.z;
  xT[((size_t)b * kN + nn) * kC + c] = f2bf(x[((size_t)b * kC + c) * kN + nn]);
}

// ---------------- kernel 3: fused QKV projection ----------------------------
// Q -> [B,H,N,DK] (scaled by log2e/sqrt(DK)), K -> [B,H,N,DK], V -> [B,H,DK,N]
__global__ __launch_bounds__(128) void attn_qkv(
    const unsigned short* __restrict__ wqb, const unsigned short* __restrict__ wkb,
    const unsigned short* __restrict__ wvb,
    const float* __restrict__ bq, const float* __restrict__ bk,
    const float* __restrict__ bv,
    const unsigned short* __restrict__ xT,
    unsigned short* __restrict__ qo, unsigned short* __restrict__ ko,
    unsigned short* __restrict__ vo) {
  const int lane  = threadIdx.x;
  const int row   = lane & 15;
  const int half  = lane >> 4;
  const int ntile = blockIdx.x * 4 + threadIdx.y;
  const int otile = blockIdx.y;
  const int b     = blockIdx.z;
  const int n     = ntile * 16 + row;

  const f32x8 fz = {0.f, 0.f, 0.f, 0.f, 0.f, 0.f, 0.f, 0.f};
  f32x8 cq = fz, ck = fz, cv = fz;
  const unsigned short* xrow = xT + ((size_t)b * kN + n) * kC + half * 16;
  const size_t arow = (size_t)(otile * 16 + row) * kC + half * 16;

#pragma unroll 2
  for (int k0 = 0; k0 < kC; k0 += 32) {
    bf16x16 bf = load_frag_b(xrow + k0);
    cq = wmma_bf16(load_frag_a(wqb + arow + k0), bf, cq);
    ck = wmma_bf16(load_frag_a(wkb + arow + k0), bf, ck);
    cv = wmma_bf16(load_frag_a(wvb + arow + k0), bf, cv);
  }

  const int obase = otile * 16 + half * 8;     // 8 contiguous output channels
  const int h     = obase >> 6;
  const int dbase = obase & 63;
  const float qscale = 0.1803368801111204f;    // (1/sqrt(64)) * log2(e)

  float qv[8], kv[8];
#pragma unroll
  for (int r = 0; r < 8; ++r) {
    qv[r] = (cq[r] + bq[obase + r]) * qscale;
    kv[r] =  ck[r] + bk[obase + r];
  }
  const uint4 qpk = make_uint4(pack_bf2(qv[0], qv[1]), pack_bf2(qv[2], qv[3]),
                               pack_bf2(qv[4], qv[5]), pack_bf2(qv[6], qv[7]));
  const uint4 kpk = make_uint4(pack_bf2(kv[0], kv[1]), pack_bf2(kv[2], kv[3]),
                               pack_bf2(kv[4], kv[5]), pack_bf2(kv[6], kv[7]));
  const size_t qoff = (((size_t)b * kH + h) * kN + n) * kD + dbase;
  *reinterpret_cast<uint4*>(qo + qoff) = qpk;
  *reinterpret_cast<uint4*>(ko + qoff) = kpk;
#pragma unroll
  for (int r = 0; r < 8; ++r)
    vo[(((size_t)b * kH + h) * kD + dbase + r) * kN + n] =
        f2bf(cv[r] + bv[obase + r]);
}

// ---------------- kernel 4: flash attention ---------------------------------
// One wave handles 32 columns n (two 16-wide tiles) so every K/V A-fragment
// feeds two WMMAs; loops m over N in steps of 32 with prefetch of next block.
__global__ __launch_bounds__(256) void attn_fa(
    const unsigned short* __restrict__ qb, const unsigned short* __restrict__ kb,
    const unsigned short* __restrict__ vb, unsigned short* __restrict__ ob) {
  __shared__ unsigned short lds[8 * 2 * 16 * 32];   // 2 KB strip per wave
  const int lane  = threadIdx.x;
  const int w     = threadIdx.y;
  const int row   = lane & 15;
  const int half  = lane >> 4;
  const int npair = blockIdx.x * 8 + w;             // 0..kN/32-1
  const int h     = blockIdx.y;
  const int b     = blockIdx.z;
  const size_t head = ((size_t)b * kH + h) * (size_t)kN * kD;
  const unsigned short* qh = qb + head;
  const unsigned short* kh = kb + head;
  const unsigned short* vh = vb + head;             // [DK, N] layout
  const int n0 = npair * 32 + row;                  // n-tile 0 column
  const int n1 = n0 + 16;                           // n-tile 1 column

  // Q B-fragments: reused for every m-step (columns n, contraction over d)
  bf16x16 qf[2][2];
#pragma unroll
  for (int nt = 0; nt < 2; ++nt) {
    const size_t qrow = (size_t)(nt ? n1 : n0) * kD + half * 16;
    qf[nt][0] = load_frag_b(qh + qrow);
    qf[nt][1] = load_frag_b(qh + qrow + 32);
  }

  const f32x8 fz = {0.f, 0.f, 0.f, 0.f, 0.f, 0.f, 0.f, 0.f};
  f32x8 acc[2][4];
#pragma unroll
  for (int nt = 0; nt < 2; ++nt)
#pragma unroll
    for (int i = 0; i < 4; ++i) acc[nt][i] = fz;
  float mrun[2] = {-1e30f, -1e30f}, lrun[2] = {0.f, 0.f};

  for (int m0 = 0; m0 < kN; m0 += 32) {
    if (m0 + 32 < kN) {  // prefetch next m-block of K and V into cache
      __builtin_prefetch(kh + (size_t)(m0 + 32 + row) * kD, 0, 1);
      __builtin_prefetch(vh + (size_t)row * kN + m0 + 32, 0, 1);
    }
    // S' = K^T Q : each K fragment shared by both n-tiles
    f32x8 s[2][2];
#pragma unroll
    for (int t = 0; t < 2; ++t) {
      const unsigned short* krow =
          kh + (size_t)(m0 + t * 16 + row) * kD + half * 16;
      const bf16x16 k0 = load_frag_a(krow);
      const bf16x16 k1 = load_frag_a(krow + 32);
#pragma unroll
      for (int nt = 0; nt < 2; ++nt) {
        f32x8 c = fz;
        c = wmma_bf16(k0, qf[nt][0], c);
        c = wmma_bf16(k1, qf[nt][1], c);
        s[nt][t] = c;
      }
    }
    // online softmax per n-tile (logits already scaled by log2e/sqrt(DK))
    bf16x16 pf[2];
#pragma unroll
    for (int nt = 0; nt < 2; ++nt) {
      float mloc = s[nt][0][0];
#pragma unroll
      for (int t = 0; t < 2; ++t)
#pragma unroll
        for (int r = 0; r < 8; ++r) mloc = fmaxf(mloc, s[nt][t][r]);
      mloc = fmaxf(mloc, __shfl_xor(mloc, 16, 32));
      const float mnew = fmaxf(mrun[nt], mloc);
      const float corr = exp2f(mrun[nt] - mnew);
      float lsum = 0.f;
      uint4 pk[2];
#pragma unroll
      for (int t = 0; t < 2; ++t) {
        float p[8];
#pragma unroll
        for (int r = 0; r < 8; ++r) {
          p[r] = exp2f(s[nt][t][r] - mnew);
          lsum += p[r];
        }
        pk[t] = make_uint4(pack_bf2(p[0], p[1]), pack_bf2(p[2], p[3]),
                           pack_bf2(p[4], p[5]), pack_bf2(p[6], p[7]));
      }
      lsum += __shfl_xor(lsum, 16, 32);
      lrun[nt] = lrun[nt] * corr + lsum;
      mrun[nt] = mnew;
#pragma unroll
      for (int dt = 0; dt < 4; ++dt)
#pragma unroll
        for (int r = 0; r < 8; ++r) acc[nt][dt][r] *= corr;
      // bounce P' through wave-private LDS: D layout -> B-fragment layout
      unsigned short* lw = lds + (((w * 2 + nt) * 16) + row) * 32;
      *reinterpret_cast<uint4*>(lw + half * 8)      = pk[0];
      *reinterpret_cast<uint4*>(lw + 16 + half * 8) = pk[1];
      pf[nt] = load_frag_b(lw + half * 16);
    }
    // O += V * P' : each V fragment shared by both n-tiles
#pragma unroll
    for (int dt = 0; dt < 4; ++dt) {
      const bf16x16 vf =
          load_frag_a(vh + (size_t)(dt * 16 + row) * kN + m0 + half * 16);
#pragma unroll
      for (int nt = 0; nt < 2; ++nt)
        acc[nt][dt] = wmma_bf16(vf, pf[nt], acc[nt][dt]);
    }
  }

#pragma unroll
  for (int nt = 0; nt < 2; ++nt) {
    const float inv = 1.0f / lrun[nt];
    unsigned short* orow =
        ob + ((size_t)b * kN + (nt ? n1 : n0)) * kC + h * kD;
#pragma unroll
    for (int dt = 0; dt < 4; ++dt) {
      float o[8];
#pragma unroll
      for (int r = 0; r < 8; ++r) o[r] = acc[nt][dt][r] * inv;
      const uint4 pkd = make_uint4(pack_bf2(o[0], o[1]), pack_bf2(o[2], o[3]),
                                   pack_bf2(o[4], o[5]), pack_bf2(o[6], o[7]));
      *reinterpret_cast<uint4*>(orow + dt * 16 + half * 8) = pkd;
    }
  }
}

// ---------------- kernel 5: output projection + bias + residual -------------
__global__ __launch_bounds__(128) void attn_out(
    const unsigned short* __restrict__ wpb, const float* __restrict__ bp,
    const unsigned short* __restrict__ obf, const float* __restrict__ x,
    float* __restrict__ y) {
  const int lane  = threadIdx.x;
  const int row   = lane & 15;
  const int half  = lane >> 4;
  const int ntile = blockIdx.x * 4 + threadIdx.y;
  const int otile = blockIdx.y;
  const int b     = blockIdx.z;
  const int n     = ntile * 16 + row;
  const f32x8 fz = {0.f, 0.f, 0.f, 0.f, 0.f, 0.f, 0.f, 0.f};
  f32x8 c = fz;
  const unsigned short* orow = obf + ((size_t)b * kN + n) * kC + half * 16;
  const size_t arow = (size_t)(otile * 16 + row) * kC + half * 16;
#pragma unroll 2
  for (int k0 = 0; k0 < kC; k0 += 32)
    c = wmma_bf16(load_frag_a(wpb + arow + k0), load_frag_b(orow + k0), c);
  const int obase = otile * 16 + half * 8;
#pragma unroll
  for (int r = 0; r < 8; ++r) {
    const int o = obase + r;
    const size_t idx = ((size_t)b * kC + o) * kN + n;
    y[idx] = c[r] + bp[o] + x[idx];
  }
}

// ---------------- launch ----------------------------------------------------
extern "C" void kernel_launch(void* const* d_in, const int* in_sizes, int n_in,
                              void* d_out, int out_size, void* d_ws,
                              size_t ws_size, hipStream_t stream) {
  (void)in_sizes; (void)n_in; (void)out_size; (void)ws_size;
  const float* x  = (const float*)d_in[0];
  const float* wq = (const float*)d_in[1];
  const float* bq = (const float*)d_in[2];
  const float* wk = (const float*)d_in[3];
  const float* bk = (const float*)d_in[4];
  const float* wv = (const float*)d_in[5];
  const float* bv = (const float*)d_in[6];
  const float* wp = (const float*)d_in[7];
  const float* bp = (const float*)d_in[8];
  float* y = (float*)d_out;

  // workspace layout (ushort elements)
  unsigned short* ws = (unsigned short*)d_ws;
  constexpr size_t SZW  = (size_t)kC * kC;            // one weight matrix
  constexpr size_t SZX  = (size_t)kB * kN * kC;       // xT / o_bf
  constexpr size_t SZQ  = (size_t)kB * kH * kN * kD;  // q/k/v (== SZX)
  unsigned short* w_bf = ws;                     // 4 weight matrices
  unsigned short* xT   = ws + 4 * SZW;
  unsigned short* qbf  = xT + SZX;
  unsigned short* kbf  = qbf + SZQ;
  unsigned short* vbf  = kbf + SZQ;
  unsigned short* obf  = vbf + SZQ;              // total ~42.5 MB

  attn_cvt_w<<<dim3((kC * kC) / 256), dim3(256), 0, stream>>>(wq, wk, wv, wp,
                                                              w_bf);
  attn_cvt_x<<<dim3(kN / 32, kC / 8, kB), dim3(32, 8), 0, stream>>>(x, xT);
  attn_qkv<<<dim3(kN / 64, kC / 16, kB), dim3(32, 4), 0, stream>>>(
      w_bf, w_bf + SZW, w_bf + 2 * SZW, bq, bk, bv, xT, qbf, kbf, vbf);
  attn_fa<<<dim3(kN / 256, kH, kB), dim3(32, 8), 0, stream>>>(qbf, kbf, vbf,
                                                              obf);
  attn_out<<<dim3(kN / 64, kC / 16, kB), dim3(32, 4), 0, stream>>>(
      w_bf + 3 * SZW, bp, obf, x, y);
}
